// ChannelAttentionModule_20057497272359
// MI455X (gfx1250) — matrix operand order repaired and checked by reading, not compiled, not run on hardware
//
#include <hip/hip_runtime.h>

typedef __attribute__((ext_vector_type(16))) _Float16 v16h;
typedef __attribute__((ext_vector_type(8)))  _Float16 v8h;
typedef __attribute__((ext_vector_type(8)))  float    v8f;

#define B_  16
#define C_  512
#define N_  1024   // H*W = 32*32

// ---------------------------------------------------------------------------
// Kernel 0: convert x (f32, (C,N) per batch) -> Xh f16 (C,N) and Xt f16 (N,C)
// LDS-tiled 64x64 transpose so both outputs are written coalesced.
// ---------------------------------------------------------------------------
__global__ void k_convert(const float* __restrict__ x,
                          _Float16* __restrict__ Xh,
                          _Float16* __restrict__ Xt) {
  __shared__ _Float16 tile[64][65];
  const int b  = blockIdx.z;
  const int c0 = blockIdx.y * 64;
  const int n0 = blockIdx.x * 64;
  const float*    xb  = x  + (size_t)b * C_ * N_;
  _Float16*       Xhb = Xh + (size_t)b * C_ * N_;
  _Float16*       Xtb = Xt + (size_t)b * N_ * C_;

  for (int e = 0; e < 16; ++e) {
    int idx = e * 256 + threadIdx.x;
    int r = idx >> 6, col = idx & 63;          // r = c-local, col = n-local
    float v = xb[(size_t)(c0 + r) * N_ + (n0 + col)];
    _Float16 h = (_Float16)v;
    Xhb[(size_t)(c0 + r) * N_ + (n0 + col)] = h;
    tile[r][col] = h;
  }
  __syncthreads();
  for (int e = 0; e < 16; ++e) {
    int idx = e * 256 + threadIdx.x;
    int r = idx >> 6, col = idx & 63;          // r = n-local, col = c-local
    Xtb[(size_t)(n0 + r) * C_ + (c0 + col)] = tile[col][r];
  }
}

// ---------------------------------------------------------------------------
// Kernel 1: energy strip (16 rows i, all 1024 j) via WMMA f16 -> f32,
// fused row-softmax in LDS, store At[j][i] = softmax(energy)[i][j] as f16
// with a coalesced LDS transpose (one 32B v16h store per attention column).
// A[m,k] = Xt[i0+m][k];  B[k,n] = Xt[j][k]  (both contiguous per lane).
// 8 waves; wave w owns j in [w*128, w*128+128).
// __launch_bounds__(256,1): ~160 live VGPRs (8 acc tiles + 8 staged B frags)
// must NOT spill — occupancy is LDS-bound anyway (64 KB strip).
// ---------------------------------------------------------------------------
__global__ void __launch_bounds__(256, 1)
k_energy_softmax(const _Float16* __restrict__ Xt,
                 _Float16* __restrict__ At) {
  __shared__ float E[16][N_];                  // 64 KB strip of exp(energy)
  __shared__ float inv_s[16];                  // per-row 1/sum
  const int b    = blockIdx.y;
  const int i0   = blockIdx.x * 16;
  const int tid  = threadIdx.x;
  const int w    = tid >> 5;
  const int lane = tid & 31;
  const int ln   = lane & 15;
  const int hi   = lane >> 4;
  const _Float16* Xtb = Xt + (size_t)b * N_ * C_;

  v8f acc[8];
  const v8f zero = {0.f,0.f,0.f,0.f,0.f,0.f,0.f,0.f};
  for (int t = 0; t < 8; ++t) acc[t] = zero;

  union AB { v16h v; v8h h[2]; };
  const _Float16* arow  = Xtb + (size_t)(i0 + ln) * C_;
  const _Float16* brow0 = Xtb + (size_t)(w * 128 + ln) * C_;

#pragma unroll
  for (int k = 0; k < 16; ++k) {               // K = 512 = 16 * 32
    AB A;
    A.h[0] = *(const v8h*)(arow + k * 32 + hi * 16);
    A.h[1] = *(const v8h*)(arow + k * 32 + hi * 16 + 8);
    // stage all 8 B fragments, then burst the WMMAs (one clause of loads,
    // WMMAs drain with descending s_wait_loadcnt)
    v16h Bv[8];
#pragma unroll
    for (int jt = 0; jt < 8; ++jt)
      Bv[jt] = *(const v16h*)(brow0 + (size_t)(jt * 16) * C_ + k * 32 + hi * 16);
#pragma unroll
    for (int jt = 0; jt < 8; ++jt)
      acc[jt] = __builtin_amdgcn_wmma_f32_16x16x32_f16(
          false, A.v, false, Bv[jt], (short)0, acc[jt], false, false);
  }
  // D layout: row m = vr + 8*hi, col n = ln
#pragma unroll
  for (int jt = 0; jt < 8; ++jt)
#pragma unroll
    for (int vr = 0; vr < 8; ++vr)
      E[vr + 8 * hi][w * 128 + jt * 16 + ln] = acc[jt][vr];
  __syncthreads();

  // row softmax reductions: 2 rows per wave; exp kept in LDS, inv in shared
  for (int q = 0; q < 2; ++q) {
    int il = w * 2 + q;
    float mx = -3.4e38f;
    for (int t = 0; t < 32; ++t) mx = fmaxf(mx, E[il][lane + 32 * t]);
    for (int o = 16; o > 0; o >>= 1) mx = fmaxf(mx, __shfl_xor(mx, o, 32));
    float s = 0.f;
    for (int t = 0; t < 32; ++t) {
      float e = __expf(E[il][lane + 32 * t] - mx);
      E[il][lane + 32 * t] = e;
      s += e;
    }
    for (int o = 16; o > 0; o >>= 1) s += __shfl_xor(s, o, 32);
    if (lane == 0) inv_s[il] = 1.f / s;
  }
  __syncthreads();

  // transposed store through LDS: thread handles column j, writes 16 halfs
  _Float16* Atb = At + (size_t)b * N_ * N_;
  for (int t = 0; t < 4; ++t) {
    int j = t * 256 + tid;
    union { v16h v; _Float16 e[16]; } u;
#pragma unroll
    for (int i = 0; i < 16; ++i)
      u.e[i] = (_Float16)(E[i][j] * inv_s[i]);
    *(v16h*)(Atb + (size_t)j * N_ + i0) = u.v;
  }
}

// ---------------------------------------------------------------------------
// Kernel 2: out[j,c] = sum_i At[j,i] * Xt[i,c], fused epilogue
//   d_out[b][j*512+c] = gamma[j*512+c] * acc + x[b][j*512+c]
// (the Keras reshape is a raw reinterpretation, so flat index matches).
// Double-buffered A/B fragments so k+1 loads overlap the k WMMA burst.
// WG tile: 128 j x 64 c; wave w owns j rows [j0+w*16, +16), 4 c-tiles.
// ---------------------------------------------------------------------------
__global__ void __launch_bounds__(256, 1)
k_out(const _Float16* __restrict__ At,
      const _Float16* __restrict__ Xh,
      const float* __restrict__ x,
      const float* __restrict__ gamma,
      float* __restrict__ out) {
  const int b    = blockIdx.z;
  const int j0   = blockIdx.x * 128;
  const int c0   = blockIdx.y * 64;
  const int tid  = threadIdx.x;
  const int w    = tid >> 5;
  const int lane = tid & 31;
  const int ln   = lane & 15;
  const int hi   = lane >> 4;
  const _Float16* Atb = At + (size_t)b * N_ * N_;
  const _Float16* Xhb = Xh + (size_t)b * C_ * N_;
  const float*    xb  = x  + (size_t)b * C_ * N_;
  float*          ob  = out + (size_t)b * C_ * N_;

  v8f acc[4];
  const v8f zero = {0.f,0.f,0.f,0.f,0.f,0.f,0.f,0.f};
  for (int t = 0; t < 4; ++t) acc[t] = zero;

  union AB { v16h v; v8h h[2]; };
  const _Float16* arow  = Atb + (size_t)(j0 + w * 16 + ln) * N_;
  const _Float16* brow0 = Xhb + (size_t)(c0 + ln) * N_;

  AB A[2];
  v16h Bv[2][4];
  A[0].h[0] = *(const v8h*)(arow + hi * 16);
  A[0].h[1] = *(const v8h*)(arow + hi * 16 + 8);
#pragma unroll
  for (int ct = 0; ct < 4; ++ct)
    Bv[0][ct] = *(const v16h*)(brow0 + (size_t)(ct * 16) * N_ + hi * 16);

#pragma unroll
  for (int k = 0; k < 32; ++k) {               // K = 1024 = 32 * 32
    const int cur = k & 1, nxt = cur ^ 1;
    if (k < 31) {
      A[nxt].h[0] = *(const v8h*)(arow + (k + 1) * 32 + hi * 16);
      A[nxt].h[1] = *(const v8h*)(arow + (k + 1) * 32 + hi * 16 + 8);
#pragma unroll
      for (int ct = 0; ct < 4; ++ct)
        Bv[nxt][ct] = *(const v16h*)(brow0 + (size_t)(ct * 16) * N_ +
                                     (k + 1) * 32 + hi * 16);
    }
#pragma unroll
    for (int ct = 0; ct < 4; ++ct)
      acc[ct] = __builtin_amdgcn_wmma_f32_16x16x32_f16(
          false, A[cur].v, false, Bv[cur][ct], (short)0, acc[ct], false, false);
  }

#pragma unroll
  for (int ct = 0; ct < 4; ++ct)
#pragma unroll
    for (int vr = 0; vr < 8; ++vr) {
      int j = j0 + w * 16 + vr + 8 * hi;
      int c = c0 + ct * 16 + ln;
      size_t kk = (size_t)j * C_ + c;
      ob[kk] = gamma[kk] * acc[ct][vr] + xb[kk];
    }
}

// ---------------------------------------------------------------------------
extern "C" void kernel_launch(void* const* d_in, const int* in_sizes, int n_in,
                              void* d_out, int out_size, void* d_ws, size_t ws_size,
                              hipStream_t stream) {
  const float* x     = (const float*)d_in[0];   // (B,C,H,W) f32
  const float* gamma = (const float*)d_in[1];   // (C,H,W)   f32
  float*       out   = (float*)d_out;           // (B,C,H,W) f32

  char* ws = (char*)d_ws;
  _Float16* Xt = (_Float16*)(ws);                       // 16 MB  (B,N,C) f16
  _Float16* Xh = (_Float16*)(ws + (16u << 20));         // 16 MB  (B,C,N) f16
  _Float16* At = (_Float16*)(ws + (32u << 20));         // 32 MB  (B,N,N) f16

  k_convert<<<dim3(N_ / 64, C_ / 64, B_), 256, 0, stream>>>(x, Xh, Xt);
  k_energy_softmax<<<dim3(N_ / 16, B_), 256, 0, stream>>>(Xt, At);
  k_out<<<dim3(N_ / 128, C_ / 64, B_), 256, 0, stream>>>(At, Xh, x, gamma, out);
}
